// TransformerBlock_32306744001019
// MI455X (gfx1250) — matrix operand order repaired and checked by reading, not compiled, not run on hardware
//
#include <hip/hip_runtime.h>

// ---------------------------------------------------------------------------
// Transformer block forward for MI455X (gfx1250, wave32, WMMA bf16,
// async global->LDS copies, double-buffered GEMM pipeline).
// ---------------------------------------------------------------------------

#define BB 8
#define SS 1024
#define DD 768
#define HH 12
#define HD 64
#define D4 3072
#define BS (BB * SS)

typedef __attribute__((ext_vector_type(16))) __bf16          v16bf;
typedef __attribute__((ext_vector_type(8)))  __bf16          v8bf;
typedef __attribute__((ext_vector_type(8)))  float           v8f;
typedef __attribute__((ext_vector_type(8)))  unsigned short  v8us;

union Frag16 { v16bf v; v8us h[2]; };

static __device__ __forceinline__ unsigned short f32bf(float f) {
  union { float f; unsigned u; } c; c.f = f;
  unsigned u = c.u;
  u += 0x7FFFu + ((u >> 16) & 1u);   // round-to-nearest-even
  return (unsigned short)(u >> 16);
}

static __device__ __forceinline__ float gelu_tanh(float x) {
  const float c = 0.79788456080286535588f; // sqrt(2/pi)
  return 0.5f * x * (1.0f + tanhf(c * (x + 0.044715f * x * x * x)));
}

static __device__ __forceinline__ float rmax16(float v) {
  v = fmaxf(v, __shfl_xor(v, 1, 32));
  v = fmaxf(v, __shfl_xor(v, 2, 32));
  v = fmaxf(v, __shfl_xor(v, 4, 32));
  v = fmaxf(v, __shfl_xor(v, 8, 32));
  return v;
}
static __device__ __forceinline__ float rsum16(float v) {
  v += __shfl_xor(v, 1, 32);
  v += __shfl_xor(v, 2, 32);
  v += __shfl_xor(v, 4, 32);
  v += __shfl_xor(v, 8, 32);
  return v;
}

// Async global->LDS copy (CDNA5, tracked by ASYNCcnt). ldsAddr is the low
// 32 bits of the flat LDS pointer (== LDS byte address per aperture rules).
static __device__ __forceinline__ void async_load_b128(unsigned ldsAddr,
                                                       const void* gaddr) {
  asm volatile("global_load_async_to_lds_b128 %0, %1, off"
               :: "v"(ldsAddr), "v"(gaddr) : "memory");
}
static __device__ __forceinline__ void wait_asynccnt0() {
  asm volatile("s_wait_asynccnt 0x0" ::: "memory");
}

// ---------------------------------------------------------------------------
// Weight transpose + f32 -> bf16:  Wt[n][k] = (bf16) W[k][n]
// ---------------------------------------------------------------------------
__global__ void transpose_bf16_kernel(const float* __restrict__ W,
                                      unsigned short* __restrict__ Wt,
                                      int K, int N) {
  size_t total = (size_t)K * N;
  for (size_t idx = (size_t)blockIdx.x * blockDim.x + threadIdx.x;
       idx < total; idx += (size_t)gridDim.x * blockDim.x) {
    size_t n = idx / (size_t)K;
    size_t k = idx - n * (size_t)K;
    Wt[idx] = f32bf(W[k * (size_t)N + n]);
  }
}

// ---------------------------------------------------------------------------
// LayerNorm (one block per row of 768) -> bf16 output
// ---------------------------------------------------------------------------
__global__ __launch_bounds__(256) void ln_bf16_kernel(
    const float* __restrict__ x, const float* __restrict__ g,
    const float* __restrict__ sh, unsigned short* __restrict__ out) {
  __shared__ float redS[256];
  __shared__ float redQ[256];
  int row = blockIdx.x, tid = threadIdx.x;
  const float* xr = x + (size_t)row * DD;
  float a = xr[tid], b = xr[tid + 256], c = xr[tid + 512];
  redS[tid] = a + b + c;
  redQ[tid] = a * a + b * b + c * c;
  __syncthreads();
  for (int o = 128; o > 0; o >>= 1) {
    if (tid < o) { redS[tid] += redS[tid + o]; redQ[tid] += redQ[tid + o]; }
    __syncthreads();
  }
  float mean = redS[0] * (1.0f / DD);
  float var  = redQ[0] * (1.0f / DD) - mean * mean;
  float rstd = rsqrtf(var + 1e-5f);
  unsigned short* orow = out + (size_t)row * DD;
  orow[tid]       = f32bf(g[tid]       * (a - mean) * rstd + sh[tid]);
  orow[tid + 256] = f32bf(g[tid + 256] * (b - mean) * rstd + sh[tid + 256]);
  orow[tid + 512] = f32bf(g[tid + 512] * (c - mean) * rstd + sh[tid + 512]);
}

// ---------------------------------------------------------------------------
// Generic bf16 GEMM:  C[M,N] = A[M,K] * W[K,N]  (W pre-transposed: Wt[N][K])
// Requires K % 64 == 0. Block tile 64x64, 8 waves, each wave computes 16x32.
// Double-buffered LDS (compile-time buffer indices, K-loop unrolled by 2)
// filled with global_load_async_to_lds_b128; one barrier per 32-K step.
// Fused epilogue: +bias, GELU, +residual(f32), f32/bf16 stores.
// ---------------------------------------------------------------------------
#define GLDA 40  // padded LDS row stride (halves), 80B = multiple of 16B

__global__ __launch_bounds__(256) void gemm_bf16_kernel(
    const unsigned short* __restrict__ A, const unsigned short* __restrict__ Wt,
    const float* __restrict__ bias, const float* __restrict__ residual,
    float* __restrict__ outF, unsigned short* __restrict__ outB,
    int M, int N, int K, int act) {
  __shared__ unsigned short As[2][64 * GLDA];
  __shared__ unsigned short Bs[2][64 * GLDA];

  const int tid   = threadIdx.x;
  const int lane  = tid & 31, wave = tid >> 5;
  const int hi    = lane >> 4, lo = lane & 15;
  const int mt    = wave & 3;       // 0..3  -> M subtile
  const int npair = wave >> 2;      // 0..1  -> pair of N subtiles
  const int mBase = blockIdx.y * 64, nBase = blockIdx.x * 64;

  // cooperative tile loads: each thread moves one 16B chunk per tile per buf
  const int lr = tid >> 2;          // 0..63 row in tile
  const int lc = (tid & 3) * 8;     // 0,8,16,24 within k-step
  const unsigned short* Ap = A  + (size_t)(mBase + lr) * K + lc;
  const unsigned short* Bp = Wt + (size_t)(nBase + lr) * K + lc;
  const unsigned aDst0 = (unsigned)(size_t)&As[0][lr * GLDA + lc];
  const unsigned aDst1 = (unsigned)(size_t)&As[1][lr * GLDA + lc];
  const unsigned bDst0 = (unsigned)(size_t)&Bs[0][lr * GLDA + lc];
  const unsigned bDst1 = (unsigned)(size_t)&Bs[1][lr * GLDA + lc];

  // fragment read offsets (constant across the K loop)
  const int arow = (mt * 16 + lo) * GLDA;
  const int n0r  = (npair * 32 + lo) * GLDA;
  const int n1r  = n0r + 16 * GLDA;

  v8f c0 = {0.f, 0.f, 0.f, 0.f, 0.f, 0.f, 0.f, 0.f};
  v8f c1 = {0.f, 0.f, 0.f, 0.f, 0.f, 0.f, 0.f, 0.f};

#define GEMM_STEP(BUF)                                                        \
  {                                                                           \
    Frag16 a, b0, b1;                                                         \
    a.h[0]  = *(const v8us*)&As[BUF][arow + 8 * hi];                          \
    a.h[1]  = *(const v8us*)&As[BUF][arow + 16 + 8 * hi];                     \
    b0.h[0] = *(const v8us*)&Bs[BUF][n0r + 8 * hi];                           \
    b0.h[1] = *(const v8us*)&Bs[BUF][n0r + 16 + 8 * hi];                      \
    b1.h[0] = *(const v8us*)&Bs[BUF][n1r + 8 * hi];                           \
    b1.h[1] = *(const v8us*)&Bs[BUF][n1r + 16 + 8 * hi];                      \
    c0 = __builtin_amdgcn_wmma_f32_16x16x32_bf16(false, a.v, false, b0.v,     \
                                                 (short)0, c0, false, false); \
    c1 = __builtin_amdgcn_wmma_f32_16x16x32_bf16(false, a.v, false, b1.v,     \
                                                 (short)0, c1, false, false); \
  }

  // prologue: fill buffer 0 with k-step 0
  async_load_b128(aDst0, Ap);
  async_load_b128(bDst0, Bp);
  wait_asynccnt0();
  __syncthreads();

  for (int k0 = 0; k0 < K; k0 += 64) {
    // K % 64 == 0 -> k0+32 always in range: fill buffer 1
    async_load_b128(aDst1, Ap + k0 + 32);
    async_load_b128(bDst1, Bp + k0 + 32);
    GEMM_STEP(0)
    wait_asynccnt0();
    __syncthreads();

    if (k0 + 64 < K) {               // refill buffer 0 for the next pass
      async_load_b128(aDst0, Ap + k0 + 64);
      async_load_b128(bDst0, Bp + k0 + 64);
    }
    GEMM_STEP(1)
    wait_asynccnt0();
    __syncthreads();
  }
#undef GEMM_STEP

  // epilogue: C rows m = r + 8*hi, cols = lane%16 within each 16-wide subtile
  const int n0 = nBase + npair * 32 + lo;
  const int n1 = n0 + 16;
  const float bia0 = bias ? bias[n0] : 0.0f;
  const float bia1 = bias ? bias[n1] : 0.0f;
#pragma unroll
  for (int r = 0; r < 8; ++r) {
    const int m = mBase + mt * 16 + r + 8 * hi;
    float v0 = c0[r] + bia0;
    float v1 = c1[r] + bia1;
    if (act) { v0 = gelu_tanh(v0); v1 = gelu_tanh(v1); }
    const size_t i0 = (size_t)m * N + n0;
    const size_t i1 = (size_t)m * N + n1;
    if (residual) { v0 += residual[i0]; v1 += residual[i1]; }
    if (outF) { outF[i0] = v0; outF[i1] = v1; }
    if (outB) { outB[i0] = f32bf(v0); outB[i1] = f32bf(v1); }
  }
}

// ---------------------------------------------------------------------------
// Flash attention, causal, scale = D**0.25 (faithful reference quirk).
// One WG per (b, h, 128-query chunk). Entire K [1024x64] (async-copied) and
// Vt [64x1024] staged in LDS (~290 KB of the 320 KB WGP LDS).
// ---------------------------------------------------------------------------
#define KP 72    // Ksh row stride (halves)
#define VP 1032  // Vtsh row stride (halves)
#define PP 40    // P tile row stride (halves)
#define ATTN_SMEM ((SS * KP + HD * VP + 8 * 16 * PP) * 2)

__global__ __launch_bounds__(256) void attn_flash_kernel(
    const unsigned short* __restrict__ qg, const unsigned short* __restrict__ kg,
    const unsigned short* __restrict__ vg, unsigned short* __restrict__ ctx) {
  extern __shared__ char smemRaw[];
  unsigned short* Ksh  = (unsigned short*)smemRaw;     // [1024][72]
  unsigned short* Vtsh = Ksh + SS * KP;                // [64][1032] (transposed)
  unsigned short* Psh  = Vtsh + HD * VP;               // [8 waves][16][40]

  const int qc = blockIdx.x, h = blockIdx.y, b = blockIdx.z;
  const int tid = threadIdx.x, lane = tid & 31, wave = tid >> 5;
  const int hi = lane >> 4, lo = lane & 15;
  const float SCALE = 0.18995915f;  // 1 / 768**0.25

  const size_t baseBH = ((size_t)b * SS) * DD + (size_t)h * HD;

  // ---- stage K (async copy) and V (transposed, VALU) into LDS ----
  for (int cc = tid; cc < SS * 8; cc += 256) {
    const int s2 = cc >> 3;
    const int ch = (cc & 7) * 8;
    async_load_b128((unsigned)(size_t)&Ksh[s2 * KP + ch],
                    &kg[baseBH + (size_t)s2 * DD + ch]);
    v8us vv = *(const v8us*)&vg[baseBH + (size_t)s2 * DD + ch];
#pragma unroll
    for (int j = 0; j < 8; ++j) Vtsh[(ch + j) * VP + s2] = vv[j];
  }
  wait_asynccnt0();
  __syncthreads();

  const int qbase = qc * 128 + wave * 16;

  // ---- Q A-fragments (K-dim = 64 -> two 16x32 fragments) ----
  Frag16 a0, a1;
  {
    const size_t qrow = baseBH + (size_t)(qbase + lo) * DD;
    a0.h[0] = *(const v8us*)&qg[qrow + 8 * hi];
    a0.h[1] = *(const v8us*)&qg[qrow + 16 + 8 * hi];
    a1.h[0] = *(const v8us*)&qg[qrow + 32 + 8 * hi];
    a1.h[1] = *(const v8us*)&qg[qrow + 48 + 8 * hi];
  }

  float mrow[8], lrow[8];
  v8f acc0 = {0.f,0.f,0.f,0.f,0.f,0.f,0.f,0.f};
  v8f acc1 = acc0, acc2 = acc0, acc3 = acc0;
#pragma unroll
  for (int r = 0; r < 8; ++r) { mrow[r] = -1e30f; lrow[r] = 0.0f; }

  unsigned short* Pw = Psh + wave * 16 * PP;

  for (int kb2 = 0; kb2 < qbase + 16; kb2 += 32) {
    // ---- scores: two 16x16 tiles (k-cols kb2..kb2+15, kb2+16..kb2+31) ----
    v8f sc0, sc1;
    {
      Frag16 b0, b1;
      const int srow = (kb2 + lo) * KP;
      b0.h[0] = *(const v8us*)&Ksh[srow + 8 * hi];
      b0.h[1] = *(const v8us*)&Ksh[srow + 16 + 8 * hi];
      b1.h[0] = *(const v8us*)&Ksh[srow + 32 + 8 * hi];
      b1.h[1] = *(const v8us*)&Ksh[srow + 48 + 8 * hi];
      v8f c = {0.f,0.f,0.f,0.f,0.f,0.f,0.f,0.f};
      c = __builtin_amdgcn_wmma_f32_16x16x32_bf16(false, a0.v, false, b0.v,
                                                  (short)0, c, false, false);
      c = __builtin_amdgcn_wmma_f32_16x16x32_bf16(false, a1.v, false, b1.v,
                                                  (short)0, c, false, false);
      sc0 = c;
    }
    {
      Frag16 b0, b1;
      const int srow = (kb2 + 16 + lo) * KP;
      b0.h[0] = *(const v8us*)&Ksh[srow + 8 * hi];
      b0.h[1] = *(const v8us*)&Ksh[srow + 16 + 8 * hi];
      b1.h[0] = *(const v8us*)&Ksh[srow + 32 + 8 * hi];
      b1.h[1] = *(const v8us*)&Ksh[srow + 48 + 8 * hi];
      v8f c = {0.f,0.f,0.f,0.f,0.f,0.f,0.f,0.f};
      c = __builtin_amdgcn_wmma_f32_16x16x32_bf16(false, a0.v, false, b0.v,
                                                  (short)0, c, false, false);
      c = __builtin_amdgcn_wmma_f32_16x16x32_bf16(false, a1.v, false, b1.v,
                                                  (short)0, c, false, false);
      sc1 = c;
    }

    // ---- online softmax over 32 new k columns, store P (bf16) to LDS ----
#pragma unroll
    for (int r = 0; r < 8; ++r) {
      const int qi = qbase + r + 8 * hi;
      const int k0 = kb2 + lo;
      const int k1 = kb2 + 16 + lo;
      const float s0 = (k0 <= qi) ? sc0[r] * SCALE : -1e30f;
      const float s1 = (k1 <= qi) ? sc1[r] * SCALE : -1e30f;
      float mx = rmax16(fmaxf(s0, s1));
      const float mn = fmaxf(mrow[r], mx);
      const float f  = __expf(mrow[r] - mn);
      const float p0 = (k0 <= qi) ? __expf(s0 - mn) : 0.0f;
      const float p1 = (k1 <= qi) ? __expf(s1 - mn) : 0.0f;
      lrow[r] = lrow[r] * f + rsum16(p0 + p1);
      mrow[r] = mn;
      acc0[r] *= f; acc1[r] *= f; acc2[r] *= f; acc3[r] *= f;
      const int m = r + 8 * hi;
      Pw[m * PP + lo]      = f32bf(p0);
      Pw[m * PP + 16 + lo] = f32bf(p1);
    }

    // ---- O += P(16x32) x V(32x64): 4 WMMAs (per-wave LDS, in-order DS) ----
    Frag16 ap;
    ap.h[0] = *(const v8us*)&Pw[lo * PP + 8 * hi];
    ap.h[1] = *(const v8us*)&Pw[lo * PP + 16 + 8 * hi];
#pragma unroll
    for (int dt = 0; dt < 4; ++dt) {
      Frag16 bv;
      const int vr = (dt * 16 + lo) * VP + kb2 + 8 * hi;
      bv.h[0] = *(const v8us*)&Vtsh[vr];
      bv.h[1] = *(const v8us*)&Vtsh[vr + 16];
      if (dt == 0) acc0 = __builtin_amdgcn_wmma_f32_16x16x32_bf16(false, ap.v, false, bv.v, (short)0, acc0, false, false);
      if (dt == 1) acc1 = __builtin_amdgcn_wmma_f32_16x16x32_bf16(false, ap.v, false, bv.v, (short)0, acc1, false, false);
      if (dt == 2) acc2 = __builtin_amdgcn_wmma_f32_16x16x32_bf16(false, ap.v, false, bv.v, (short)0, acc2, false, false);
      if (dt == 3) acc3 = __builtin_amdgcn_wmma_f32_16x16x32_bf16(false, ap.v, false, bv.v, (short)0, acc3, false, false);
    }
  }

  // ---- normalize + store ctx (layout [B,S,H,HD] == rows of [BS, D]) ----
#pragma unroll
  for (int r = 0; r < 8; ++r) {
    const float inv = 1.0f / lrow[r];
    const size_t orow = baseBH + (size_t)(qbase + r + 8 * hi) * DD;
    ctx[orow +  0 + lo] = f32bf(acc0[r] * inv);
    ctx[orow + 16 + lo] = f32bf(acc1[r] * inv);
    ctx[orow + 32 + lo] = f32bf(acc2[r] * inv);
    ctx[orow + 48 + lo] = f32bf(acc3[r] * inv);
  }
}

// ---------------------------------------------------------------------------
// Orchestration
// ---------------------------------------------------------------------------
extern "C" void kernel_launch(void* const* d_in, const int* in_sizes, int n_in,
                              void* d_out, int out_size, void* d_ws, size_t ws_size,
                              hipStream_t stream) {
  const float* x  = (const float*)d_in[0];
  const float* Wq = (const float*)d_in[1];
  const float* Wk = (const float*)d_in[2];
  const float* Wv = (const float*)d_in[3];
  const float* Wo = (const float*)d_in[4];
  const float* bo = (const float*)d_in[5];
  const float* W1 = (const float*)d_in[6];
  const float* b1 = (const float*)d_in[7];
  const float* W2 = (const float*)d_in[8];
  const float* b2 = (const float*)d_in[9];
  const float* g1 = (const float*)d_in[10];
  const float* s1 = (const float*)d_in[11];
  const float* g2 = (const float*)d_in[12];
  const float* s2 = (const float*)d_in[13];

  char* ws = (char*)d_ws;
  size_t off = 0;
  auto alloc = [&](size_t bytes) -> void* {
    void* p = ws + off;
    off += (bytes + 255) & ~(size_t)255;
    return p;
  };

  unsigned short* hb   = (unsigned short*)alloc((size_t)BS * DD * 2);
  unsigned short* qb   = (unsigned short*)alloc((size_t)BS * DD * 2);
  unsigned short* kb   = (unsigned short*)alloc((size_t)BS * DD * 2);
  unsigned short* vb   = (unsigned short*)alloc((size_t)BS * DD * 2);
  unsigned short* ctxb = (unsigned short*)alloc((size_t)BS * DD * 2);
  unsigned short* h2b  = (unsigned short*)alloc((size_t)BS * DD * 2);
  unsigned short* m1b  = (unsigned short*)alloc((size_t)BS * D4 * 2);
  float*          x1f  = (float*)alloc((size_t)BS * DD * 4);
  unsigned short* WqT  = (unsigned short*)alloc((size_t)DD * DD * 2);
  unsigned short* WkT  = (unsigned short*)alloc((size_t)DD * DD * 2);
  unsigned short* WvT  = (unsigned short*)alloc((size_t)DD * DD * 2);
  unsigned short* WoT  = (unsigned short*)alloc((size_t)DD * DD * 2);
  unsigned short* W1T  = (unsigned short*)alloc((size_t)DD * D4 * 2);
  unsigned short* W2T  = (unsigned short*)alloc((size_t)DD * D4 * 2);
  (void)ws_size; (void)in_sizes; (void)n_in; (void)out_size;

  const int TB = 256;
  const int nDD = (DD * DD + TB - 1) / TB;
  const int nD4 = (DD * D4 + TB - 1) / TB;
  transpose_bf16_kernel<<<nDD, TB, 0, stream>>>(Wq, WqT, DD, DD);
  transpose_bf16_kernel<<<nDD, TB, 0, stream>>>(Wk, WkT, DD, DD);
  transpose_bf16_kernel<<<nDD, TB, 0, stream>>>(Wv, WvT, DD, DD);
  transpose_bf16_kernel<<<nDD, TB, 0, stream>>>(Wo, WoT, DD, DD);
  transpose_bf16_kernel<<<nD4, TB, 0, stream>>>(W1, W1T, DD, D4);  // [3072][768]
  transpose_bf16_kernel<<<nD4, TB, 0, stream>>>(W2, W2T, D4, DD);  // [768][3072]

  // pre-LN 1
  ln_bf16_kernel<<<BS, 256, 0, stream>>>(x, g1, s1, hb);

  // QKV projections (no bias in reference)
  dim3 gQKV(DD / 64, BS / 64);
  gemm_bf16_kernel<<<gQKV, 256, 0, stream>>>(hb, WqT, nullptr, nullptr, nullptr, qb, BS, DD, DD, 0);
  gemm_bf16_kernel<<<gQKV, 256, 0, stream>>>(hb, WkT, nullptr, nullptr, nullptr, kb, BS, DD, DD, 0);
  gemm_bf16_kernel<<<gQKV, 256, 0, stream>>>(hb, WvT, nullptr, nullptr, nullptr, vb, BS, DD, DD, 0);

  // causal flash attention
  dim3 gAttn(SS / 128, HH, BB);
  attn_flash_kernel<<<gAttn, 256, ATTN_SMEM, stream>>>(qb, kb, vb, ctxb);

  // output projection + bo + residual(x) -> x1 (f32)
  gemm_bf16_kernel<<<gQKV, 256, 0, stream>>>(ctxb, WoT, bo, x, x1f, nullptr, BS, DD, DD, 0);

  // pre-LN 2
  ln_bf16_kernel<<<BS, 256, 0, stream>>>(x1f, g2, s2, h2b);

  // MLP: gelu(h2 @ W1 + b1) @ W2 + b2 + x1 -> out (f32)
  dim3 gMLP1(D4 / 64, BS / 64);
  gemm_bf16_kernel<<<gMLP1, 256, 0, stream>>>(h2b, W1T, b1, nullptr, nullptr, m1b, BS, D4, DD, 1);
  gemm_bf16_kernel<<<gQKV, 256, 0, stream>>>(m1b, W2T, b2, x1f, (float*)d_out, nullptr, BS, DD, D4, 0);
}
